// Attention_481036337480
// MI455X (gfx1250) — matrix operand order repaired and checked by reading, not compiled, not run on hardware
//
#include <hip/hip_runtime.h>
#include <hip/hip_bf16.h>
#include <cstdint>

// ---- geometry (fixed by the reference) ----
#define BATCH 4
#define SEQ   2048
#define DMODEL 512
#define NHEAD 8
#define DHEAD 64
#define NEG_INF_F (-1.0e9f)
#define INV_SCALE 0.03125f   // 1 / (dh/2) = 1/32

typedef __bf16 bf16_t;
typedef __attribute__((ext_vector_type(8)))  __bf16 v8bf;
typedef __attribute__((ext_vector_type(16))) __bf16 v16bf;
typedef __attribute__((ext_vector_type(8)))  float  v8f;

#if __has_builtin(__builtin_amdgcn_global_load_async_to_lds_b32)
#define HAVE_ASYNC_LDS 1
typedef __attribute__((address_space(1))) int g_int_t;   // global (AS1)
typedef __attribute__((address_space(3))) int l_int_t;   // LDS (AS3)
__device__ inline void wait_async0() {
#if __has_builtin(__builtin_amdgcn_s_wait_asynccnt)
    __builtin_amdgcn_s_wait_asynccnt(0);
#else
    asm volatile("s_wait_asynccnt 0x0" ::: "memory");
#endif
}
#endif

union V16 {
    v16bf v;
    v8bf  h[2];
    uint16_t u[16];
};

__device__ inline uint16_t f2bf(float x) {
    union { float f; uint32_t u; } c; c.f = x;
    uint32_t u = c.u + 0x7FFFu + ((c.u >> 16) & 1u);  // round-to-nearest-even
    return (uint16_t)(u >> 16);
}

// ---------------------------------------------------------------------------
// Projection GEMM: Y[8192,512] = X[8192,512] @ W[512,512] + b
// One workgroup (256 thr = 8 waves) computes a 128(M) x 64(N) tile.
// Each wave: 16 rows x 64 cols = 4 WMMA accumulators; K-loop step 32 (bf16).
// W tile staged transposed in LDS; on CDNA5 the transpose rides the async
// global->LDS path (per-lane scatter), then f32->bf16 conversion in LDS.
// ---------------------------------------------------------------------------
template <bool IN_F32, bool OUT_F32>
__global__ __launch_bounds__(256)
void proj_gemm(const void* __restrict__ Xv, const float* __restrict__ W,
               const float* __restrict__ bias, void* __restrict__ Yv)
{
    __shared__ __align__(32) uint16_t lw[64 * 32];   // [n][k] transposed W tile (bf16 bits)
#if defined(HAVE_ASYNC_LDS)
    __shared__ __align__(16) float lwf[64 * 32];     // async staging (f32, transposed)
#endif

    const int tid  = threadIdx.x;
    const int lane = tid & 31;
    const int wv   = tid >> 5;
    const int ln   = lane & 15;
    const int kh   = lane >> 4;

    const int m0 = blockIdx.x * 128 + wv * 16;
    const int n0 = blockIdx.y * 64;
    const int mrow = m0 + ln;

    v8f acc[4] = {};

    for (int k0 = 0; k0 < DMODEL; k0 += 32) {
#if defined(HAVE_ASYNC_LDS)
        // async DMA: W[k0+kk][n0+nn] -> lwf[nn][kk] (transpose via per-lane addrs)
        #pragma unroll
        for (int j = 0; j < 8; ++j) {
            int e  = j * 256 + tid;
            int kk = e >> 6;
            int nn = e & 63;
            const float* src = &W[(size_t)(k0 + kk) * DMODEL + (n0 + nn)];
            __builtin_amdgcn_global_load_async_to_lds_b32(
                (g_int_t*)(void*)src, (l_int_t*)&lwf[nn * 32 + kk], 0, 0);
        }
        wait_async0();
        __syncthreads();
        // cooperative f32 -> bf16 conversion in LDS
        #pragma unroll
        for (int j = 0; j < 8; ++j) {
            int e = j * 256 + tid;
            lw[e] = f2bf(lwf[e]);
        }
#else
        // cooperative load of W[k0..k0+31][n0..n0+63] -> LDS transposed [n][k]
        #pragma unroll
        for (int j = 0; j < 8; ++j) {
            int e  = j * 256 + tid;
            int kk = e >> 6;
            int nn = e & 63;
            lw[nn * 32 + kk] = f2bf(W[(size_t)(k0 + kk) * DMODEL + (n0 + nn)]);
        }
#endif
        __syncthreads();

        // A fragment (16x32): lane half kh holds K chunks {kh*8..+7, 16+kh*8..+7}
        V16 a;
        if constexpr (IN_F32) {
            const float* X = (const float*)Xv;
            const float* r = X + (size_t)mrow * DMODEL + k0;
            v8f c0 = *(const v8f*)(r + kh * 8);
            v8f c1 = *(const v8f*)(r + 16 + kh * 8);
            #pragma unroll
            for (int i = 0; i < 8; ++i) { a.u[i] = f2bf(c0[i]); a.u[8 + i] = f2bf(c1[i]); }
        } else {
            const uint16_t* X = (const uint16_t*)Xv;
            const uint16_t* r = X + (size_t)mrow * DMODEL + k0;
            a.h[0] = *(const v8bf*)(r + kh * 8);
            a.h[1] = *(const v8bf*)(r + 16 + kh * 8);
        }

        #pragma unroll
        for (int nt = 0; nt < 4; ++nt) {
            v16bf bfrag = __builtin_shufflevector(*(const v8bf*)&lw[(nt * 16 + ln) * 32 + kh * 8], *(const v8bf*)&lw[(nt * 16 + ln) * 32 + 16 + kh * 8], 0, 1, 2, 3, 4, 5, 6, 7, 8, 9, 10, 11, 12, 13, 14, 15);
            acc[nt] = __builtin_amdgcn_wmma_f32_16x16x32_bf16(
                false, a.v, false, bfrag, (short)0, acc[nt], false, false);
        }
        __syncthreads();
    }

    // epilogue: bias add + store (C layout: row = i + 8*kh, col = ln)
    #pragma unroll
    for (int nt = 0; nt < 4; ++nt) {
        int col = n0 + nt * 16 + ln;
        float bv = bias[col];
        #pragma unroll
        for (int i = 0; i < 8; ++i) {
            int row = m0 + i + 8 * kh;
            float val = acc[nt][i] + bv;
            if constexpr (OUT_F32)
                ((float*)Yv)[(size_t)row * DMODEL + col] = val;
            else
                ((uint16_t*)Yv)[(size_t)row * DMODEL + col] = f2bf(val);
        }
    }
}

// ---------------------------------------------------------------------------
// Attention core: one workgroup per (b, h, 128 q-rows); one wave per 16 rows.
// Scores are RECOMPUTED per pass (QK^T is ~free on WMMA vs a 537MB global
// round-trip; mask is 67MB shared across heads -> lives in 192MB L2):
//   Pass1: S = (q kT)/32 masked, rowmax only (no store).
//   Pass2: recompute S, e = exp(s - m) -> attn region, rowsum.
//   Pass3: p = e/sum -> attn region (final softmax), O += p @ v via WMMA.
// ---------------------------------------------------------------------------
__global__ __launch_bounds__(256)
void attention_core(const uint16_t* __restrict__ qb, const uint16_t* __restrict__ kb,
                    const uint16_t* __restrict__ vb, const int* __restrict__ mask,
                    float* __restrict__ attn, uint16_t* __restrict__ ob)
{
    __shared__ __align__(32) uint16_t lv[64 * 32];   // v tile transposed [dh][lk]
    __shared__ float rowstat[8][16];                 // per-wave row sums

    const int tid  = threadIdx.x;
    const int lane = tid & 31;
    const int wv   = tid >> 5;
    const int ln   = lane & 15;
    const int kh   = lane >> 4;

    const int bh    = blockIdx.x >> 4;        // b*NHEAD + h
    const int b     = bh >> 3;
    const int qtile = blockIdx.x & 15;
    const int q0    = qtile * 128 + wv * 16;  // head-local first q row for this wave

    const size_t headbase = (size_t)bh * SEQ * DHEAD;

    // preload q A-fragments for K(=d) chunks 0..31, 32..63
    V16 qf[2];
    #pragma unroll
    for (int dc = 0; dc < 2; ++dc) {
        const uint16_t* r = qb + headbase + (size_t)(q0 + ln) * DHEAD + dc * 32;
        qf[dc].h[0] = *(const v8bf*)(r + kh * 16);
        qf[dc].h[1] = *(const v8bf*)(r + kh * 16 + 8);
    }

    // ---------------- pass 1: rowmax only ----------------
    float rmax[8];
    #pragma unroll
    for (int i = 0; i < 8; ++i) rmax[i] = -3.0e38f;

    for (int lk0 = 0; lk0 < SEQ; lk0 += 16) {
        if (lk0 + 16 < SEQ)
            __builtin_prefetch((const void*)(kb + headbase + (size_t)(lk0 + 16 + ln) * DHEAD), 0, 1);

        v8f s = {};
        #pragma unroll
        for (int dc = 0; dc < 2; ++dc) {
            v16bf bf = *(const v16bf*)(kb + headbase + (size_t)(lk0 + ln) * DHEAD + dc * 32 + kh * 16);
            s = __builtin_amdgcn_wmma_f32_16x16x32_bf16(
                false, qf[dc].v, false, bf, (short)0, s, false, false);
        }
        int lk = lk0 + ln;
        #pragma unroll
        for (int i = 0; i < 8; ++i) {
            int lq = q0 + i + 8 * kh;                       // C layout row
            int mz = mask[((size_t)b * SEQ + lq) * SEQ + lk];
            float sv = (mz == 0) ? NEG_INF_F : s[i] * INV_SCALE;
            rmax[i] = fmaxf(rmax[i], sv);
        }
    }
    // reduce max across the 16 lanes of each half (columns of the tile)
    #pragma unroll
    for (int i = 0; i < 8; ++i) {
        float v = rmax[i];
        v = fmaxf(v, __shfl_xor(v, 1));
        v = fmaxf(v, __shfl_xor(v, 2));
        v = fmaxf(v, __shfl_xor(v, 4));
        v = fmaxf(v, __shfl_xor(v, 8));
        rmax[i] = v;                        // now rowmax for row (q0 + i + 8*kh)
    }

    // ---------------- pass 2: recompute scores, exp, store e, rowsum ----------------
    float rs[8];
    #pragma unroll
    for (int i = 0; i < 8; ++i) rs[i] = 0.0f;

    for (int lk0 = 0; lk0 < SEQ; lk0 += 16) {
        v8f s = {};
        #pragma unroll
        for (int dc = 0; dc < 2; ++dc) {
            v16bf bf = *(const v16bf*)(kb + headbase + (size_t)(lk0 + ln) * DHEAD + dc * 32 + kh * 16);
            s = __builtin_amdgcn_wmma_f32_16x16x32_bf16(
                false, qf[dc].v, false, bf, (short)0, s, false, false);
        }
        int lk = lk0 + ln;
        #pragma unroll
        for (int i = 0; i < 8; ++i) {
            int lq = q0 + i + 8 * kh;
            int mz = mask[((size_t)b * SEQ + lq) * SEQ + lk];
            float sv = (mz == 0) ? NEG_INF_F : s[i] * INV_SCALE;
            float e  = __expf(sv - rmax[i]);
            rs[i] += e;
            attn[((size_t)bh * SEQ + lq) * SEQ + lk] = e;   // unnormalized exp
        }
    }
    // reduce sums across the 16 lanes of each half, publish per-row
    #pragma unroll
    for (int i = 0; i < 8; ++i) {
        float v = rs[i];
        v += __shfl_xor(v, 1);
        v += __shfl_xor(v, 2);
        v += __shfl_xor(v, 4);
        v += __shfl_xor(v, 8);
        rs[i] = v;
    }
    if (ln == 0) {                       // lanes 0 and 16 publish rows 0-7 / 8-15
        #pragma unroll
        for (int i = 0; i < 8; ++i) rowstat[wv][kh * 8 + i] = rs[i];
    }
    __syncthreads();
    const float rinv = 1.0f / rowstat[wv][ln];   // row (q0+ln) sum, A-layout lane

    // ---------------- pass 3: normalize + O = P @ V ----------------
    float* erow = attn + ((size_t)bh * SEQ + q0 + ln) * SEQ;   // this lane's row
    v8f oacc[4] = {};
    for (int lk0 = 0; lk0 < SEQ; lk0 += 32) {
        __syncthreads();
        // cooperative transpose of v[lk0..+31][0..63] into LDS [dh][lk]
        #pragma unroll
        for (int j = 0; j < 8; ++j) {
            int e  = j * 256 + tid;
            int lk = e >> 6;
            int n  = e & 63;
            lv[n * 32 + lk] = vb[headbase + (size_t)(lk0 + lk) * DHEAD + n];
        }
        __syncthreads();

        // load e in A-frag layout, normalize, write final attn, build bf16 A frag
        V16 pa;
        v8f e0 = *(const v8f*)(erow + lk0 + kh * 8);
        v8f e1 = *(const v8f*)(erow + lk0 + 16 + kh * 8);
        #pragma unroll
        for (int i = 0; i < 8; ++i) {
            float p0 = e0[i] * rinv;
            float p1 = e1[i] * rinv;
            e0[i] = p0; e1[i] = p1;
            pa.u[i]     = f2bf(p0);
            pa.u[8 + i] = f2bf(p1);
        }
        *(v8f*)(erow + lk0 + kh * 8)      = e0;
        *(v8f*)(erow + lk0 + 16 + kh * 8) = e1;

        #pragma unroll
        for (int nt = 0; nt < 4; ++nt) {
            v16bf bf = __builtin_shufflevector(*(const v8bf*)&lv[(nt * 16 + ln) * 32 + kh * 8], *(const v8bf*)&lv[(nt * 16 + ln) * 32 + 16 + kh * 8], 0, 1, 2, 3, 4, 5, 6, 7, 8, 9, 10, 11, 12, 13, 14, 15);
            oacc[nt] = __builtin_amdgcn_wmma_f32_16x16x32_bf16(
                false, pa.v, false, bf, (short)0, oacc[nt], false, false);
        }
    }

    // store O tile (bf16) into the [B,H,L,dh]-flat workspace for the Wo GEMM
    #pragma unroll
    for (int nt = 0; nt < 4; ++nt) {
        #pragma unroll
        for (int i = 0; i < 8; ++i) {
            int row = q0 + i + 8 * kh;
            ob[headbase + (size_t)row * DHEAD + nt * 16 + ln] = f2bf(oacc[nt][i]);
        }
    }
}

// ---------------------------------------------------------------------------
extern "C" void kernel_launch(void* const* d_in, const int* in_sizes, int n_in,
                              void* d_out, int out_size, void* d_ws, size_t ws_size,
                              hipStream_t stream) {
    (void)in_sizes; (void)n_in; (void)out_size; (void)ws_size;

    const float* Q    = (const float*)d_in[0];
    const float* Kin  = (const float*)d_in[1];
    const float* Vin  = (const float*)d_in[2];
    const int*   mask = (const int*)d_in[3];
    const float* Wq   = (const float*)d_in[4];
    const float* bq   = (const float*)d_in[5];
    const float* Wk   = (const float*)d_in[6];
    const float* bk   = (const float*)d_in[7];
    const float* Wv   = (const float*)d_in[8];
    const float* bv   = (const float*)d_in[9];
    const float* Wo   = (const float*)d_in[10];
    const float* bo   = (const float*)d_in[11];

    float* out  = (float*)d_out;                                   // [4,2048,512]
    float* attn = out + (size_t)BATCH * SEQ * DMODEL;              // [4,8,2048,2048]

    const size_t ELEMS = (size_t)BATCH * SEQ * DMODEL;             // 4,194,304
    uint16_t* qb = (uint16_t*)d_ws;
    uint16_t* kb = qb + ELEMS;
    uint16_t* vb = kb + ELEMS;
    uint16_t* ob = vb + ELEMS;

    dim3 pg(64, 8), pb(256);
    proj_gemm<true, false><<<pg, pb, 0, stream>>>(Q,   Wq, bq, qb);
    proj_gemm<true, false><<<pg, pb, 0, stream>>>(Kin, Wk, bk, kb);
    proj_gemm<true, false><<<pg, pb, 0, stream>>>(Vin, Wv, bv, vb);

    attention_core<<<dim3(BATCH * NHEAD * 16), dim3(256), 0, stream>>>(
        qb, kb, vb, mask, attn, ob);

    proj_gemm<false, true><<<pg, pb, 0, stream>>>(ob, Wo, bo, out);
}
